// SelfAttention_18786186953097
// MI455X (gfx1250) — matrix operand order repaired and checked by reading, not compiled, not run on hardware
//
#include <hip/hip_runtime.h>
#include <hip/hip_bf16.h>

typedef __attribute__((ext_vector_type(16))) _Float16 v16h;
typedef __attribute__((ext_vector_type(8)))  _Float16 v8h;
typedef __attribute__((ext_vector_type(8)))  float    v8f;

// ---------------------------------------------------------------------------
// WMMA helpers (CDNA5 wave32, V_WMMA_F32_16X16X32_F16)
// ---------------------------------------------------------------------------
__device__ __forceinline__ v8f v8f_zero() {
    v8f z;
#pragma unroll
    for (int i = 0; i < 8; ++i) z[i] = 0.0f;
    return z;
}

__device__ __forceinline__ v8f wmma16(v16h a, v16h b, v8f c) {
    return __builtin_amdgcn_wmma_f32_16x16x32_f16(
        /*neg_a=*/false, a, /*neg_b=*/false, b,
        /*c_mod=*/(short)0, c, /*reuse_a=*/false, /*reuse_b=*/false);
}

// A-matrix fragment (16x32 f16, row-major [M][K], K window contiguous at base).
// ISA layout: lane(0-15) row M=lane, K = {h*8..h*8+7} U {16+h*8..+7}, h=lane>>4.
__device__ __forceinline__ v16h load_frag_a(const _Float16* base, int stride_h, int lane) {
    const int m = lane & 15;
    const int h = (lane >> 4) << 3;          // 0 or 8 (halfs)
    const _Float16* r = base + (size_t)m * stride_h;
    v8h lo = *(const v8h*)(r + h);           // K = h .. h+7
    v8h hi = *(const v8h*)(r + 16 + h);      // K = 16+h .. 16+h+7
    v16h out;
#pragma unroll
    for (int i = 0; i < 8; ++i) { out[i] = lo[i]; out[i + 8] = hi[i]; }
    return out;
}

// B-matrix fragment (32x16 f16) read from TRANSPOSED storage BT[N][K]
// (K contiguous). ISA layout: lane col N=lane&15, K = {h*8..h*8+7} U {16+h*8..+7}, h=lane>>4.
__device__ __forceinline__ v16h load_frag_bt(const _Float16* base, int stride_h,
                                             int k0, int lane) {
    const int n  = lane & 15;
    const int kb = k0 + ((lane >> 4) << 3);
    const _Float16* r = base + (size_t)n * stride_h + kb;
    v8h lo = *(const v8h*)(r);
    v8h hi = *(const v8h*)(r + 16);
    v16h out;
#pragma unroll
    for (int i = 0; i < 8; ++i) { out[i] = lo[i]; out[i + 8] = hi[i]; }
    return out;
}

// ---------------------------------------------------------------------------
// Prep kernels: weight f32->f16, x transpose to XT[b][n][c] (f16)
// ---------------------------------------------------------------------------
__global__ void cvt_f32_to_f16(const float* __restrict__ in,
                               _Float16* __restrict__ out, int n) {
    int i = blockIdx.x * 256 + threadIdx.x;
    if (i < n) out[i] = (_Float16)in[i];
}

// LDS-tiled 64x64 transpose: coalesced f32 reads, coalesced 128B f16 writes.
// Stride 66 halfs = 33 dwords (odd) -> bank-conflict-free in both phases.
__global__ void xt_kernel(const float* __restrict__ x, _Float16* __restrict__ xt) {
    __shared__ _Float16 tile[64][66];
    const int tid = threadIdx.x;
    const int hw0 = blockIdx.x * 64;
    const int c0  = blockIdx.y * 64;
    const int b   = blockIdx.z;
    {
        const int hwl = tid & 63;
        const int cb  = tid >> 6;                    // 0..3
#pragma unroll
        for (int i = 0; i < 16; ++i) {
            int cl = cb * 16 + i;
            tile[cl][hwl] =
                (_Float16)x[((size_t)b * 256 + c0 + cl) * 4096 + hw0 + hwl];
        }
    }
    __syncthreads();
    {
        const int cl = tid & 63;
        const int hb = tid >> 6;
#pragma unroll
        for (int i = 0; i < 16; ++i) {
            int hwl = hb * 16 + i;
            xt[((size_t)b * 4096 + hw0 + hwl) * 256 + c0 + cl] = tile[cl][hwl];
        }
    }
}

// ---------------------------------------------------------------------------
// Projection kernel: f = wf*x, g = pool(wg*x), h = pool(wh*x)
// grid (32 spatial tiles of 128, 16 batches), 384 threads = 12 waves.
// Wave owns 2 m-tiles x 4 n-tiles (8 accumulator tiles = 64 VGPRs): 6
// fragment loads per 8 WMMAs, A reused 4x, B reused 2x. ~130 VGPRs total,
// no spills. Tile of 128 positions = rows {2t,2t+1} -> 2x2 pooling local.
// ---------------------------------------------------------------------------
__device__ __forceinline__ const _Float16* proj_wbase(const _Float16* WF,
                                                      const _Float16* WG,
                                                      const _Float16* WH, int mt) {
    if (mt < 2) return WF + (size_t)(mt * 16) * 256;
    if (mt < 4) return WG + (size_t)((mt - 2) * 16) * 256;
    return WH + (size_t)((mt - 4) * 16) * 256;
}

__global__ void proj_kernel(const _Float16* __restrict__ XT,
                            const _Float16* __restrict__ WF,
                            const _Float16* __restrict__ WG,
                            const _Float16* __restrict__ WH,
                            _Float16* __restrict__ FT,   // [B][4096][32]
                            _Float16* __restrict__ GT,   // [B][1024][32]
                            _Float16* __restrict__ Hm) { // [B][128][1024]
    __shared__ __align__(16) _Float16 stg[160][128];     // g(32)+h(128) rows, pre-pool

    const int tid  = threadIdx.x;
    const int lane = tid & 31;
    const int w    = tid >> 5;            // 0..11
    const int t    = blockIdx.x;          // spatial tile (128 positions)
    const int b    = blockIdx.y;
    const int n0   = t * 128;
    const int half = lane >> 4, nl = lane & 15;
    const int mg   = (w % 6) * 2;         // first of this wave's 2 m-tiles
    const int ng   = (w / 6) * 4;         // first of this wave's 4 n-tiles

    const _Float16* xrow = XT + ((size_t)b * 4096 + n0 + ng * 16) * 256;

    // Base pointers, advanced by 32 halfs (one K-step) per iteration.
    const _Float16* wp0 = proj_wbase(WF, WG, WH, mg + 0);
    const _Float16* wp1 = proj_wbase(WF, WG, WH, mg + 1);
    const _Float16* xp0 = xrow;
    const _Float16* xp1 = xrow + (size_t)16 * 256;
    const _Float16* xp2 = xrow + (size_t)32 * 256;
    const _Float16* xp3 = xrow + (size_t)48 * 256;

    v8f acc[8];                           // acc[i*4+j] = tile (mg+i, ng+j)
#pragma unroll
    for (int i = 0; i < 8; ++i) acc[i] = v8f_zero();

#pragma unroll 1
    for (int k = 0; k < 8; ++k) {                        // K = 256 / 32
        v16h A0 = load_frag_a(wp0, 256, lane);
        v16h A1 = load_frag_a(wp1, 256, lane);
        v16h B0 = load_frag_bt(xp0, 256, 0, lane);
        v16h B1 = load_frag_bt(xp1, 256, 0, lane);
        v16h B2 = load_frag_bt(xp2, 256, 0, lane);
        v16h B3 = load_frag_bt(xp3, 256, 0, lane);

        acc[0] = wmma16(A0, B0, acc[0]);
        acc[1] = wmma16(A0, B1, acc[1]);
        acc[2] = wmma16(A0, B2, acc[2]);
        acc[3] = wmma16(A0, B3, acc[3]);
        acc[4] = wmma16(A1, B0, acc[4]);
        acc[5] = wmma16(A1, B1, acc[5]);
        acc[6] = wmma16(A1, B2, acc[6]);
        acc[7] = wmma16(A1, B3, acc[7]);

        wp0 += 32; wp1 += 32;
        xp0 += 32; xp1 += 32; xp2 += 32; xp3 += 32;
    }

    // Epilogue: f tiles straight to FT (b128 stores, oc contiguous in v);
    // g/h tiles staged in LDS for pooling (row = (mt-2)*16 + v + 8*half).
#pragma unroll
    for (int i = 0; i < 2; ++i) {
        int mt = mg + i;
#pragma unroll
        for (int j = 0; j < 4; ++j) {
            int nt = ng + j;
            v8f a = acc[i * 4 + j];
            if (mt < 2) {
                v8h o;
#pragma unroll
                for (int v = 0; v < 8; ++v) o[v] = (_Float16)a[v];
                int n = n0 + nt * 16 + nl;
                *(v8h*)(FT + ((size_t)b * 4096 + n) * 32 + mt * 16 + 8 * half) = o;
            } else {
#pragma unroll
                for (int v = 0; v < 8; ++v) {
                    int row = (mt - 2) * 16 + v + 8 * half;
                    stg[row][nt * 16 + nl] = (_Float16)a[v];
                }
            }
        }
    }
    __syncthreads();

    // 2x2 maxpool: local cols {2q,2q+1,64+2q,64+2q+1} -> pooled pos t*32+q
    for (int i = tid; i < 160 * 32; i += 384) {
        int row = i >> 5, q = i & 31;
        float a0 = (float)stg[row][2 * q];
        float a1 = (float)stg[row][2 * q + 1];
        float a2 = (float)stg[row][64 + 2 * q];
        float a3 = (float)stg[row][65 + 2 * q];
        float mx = fmaxf(fmaxf(a0, a1), fmaxf(a2, a3));
        int p = t * 32 + q;
        if (row < 32)
            GT[((size_t)b * 1024 + p) * 32 + row] = (_Float16)mx;         // GT[m][c]
        else
            Hm[((size_t)b * 128 + (row - 32)) * 1024 + p] = (_Float16)mx; // H[c][m]
    }
}

// ---------------------------------------------------------------------------
// Fused flash attention + output projection + residual.
// grid (64 query tiles of 64, 16 batches), 256 threads = 8 waves.
// Streaming softmax over 16 chunks of 64 keys; att accumulator kept in
// WMMA accumulator registers and rescaled per chunk. Softmax parallelized
// 4 threads/row via LDS partial max/sum.
// ---------------------------------------------------------------------------
__global__ void attn_kernel(const _Float16* __restrict__ FT,
                            const _Float16* __restrict__ GT,
                            const _Float16* __restrict__ Hm,
                            const _Float16* __restrict__ WO,  // [256][128] f16
                            const float*    __restrict__ x,
                            const float*    __restrict__ gamma_p,
                            float*          __restrict__ out) {
    __shared__ __align__(16) float    S[64][68];       // logits chunk
    __shared__ __align__(16) _Float16 P[64][72];       // exp(logits) (BT layout)
    __shared__ __align__(16) _Float16 ATT[64][136];    // att^T (BT layout)
    __shared__ float Mrun[64], Srun[64], Cfac[64], Inv[64];
    __shared__ float PM[64][4], PS[64][4];             // softmax partials

    const int tid  = threadIdx.x;
    const int lane = tid & 31;
    const int w    = tid >> 5;
    const int qt   = blockIdx.x;
    const int b    = blockIdx.y;
    const int n0   = qt * 64;
    const int n_t  = w & 3;               // query 16-tile owned by this wave
    const int grp  = w >> 2;              // 0/1: splits key tiles / channel tiles
    const int half = lane >> 4, nl = lane & 15;
    const int r    = tid & 63;            // softmax row
    const int seg  = tid >> 6;            // softmax 16-key segment (0..3)

    if (tid < 64) { Mrun[tid] = -1e30f; Srun[tid] = 0.0f; }
    __syncthreads();

    // A-fragment of the logit GEMM (f^T tile, K = C/8 = 32) — chunk-invariant
    v16h af = load_frag_a(FT + ((size_t)b * 4096 + n0 + n_t * 16) * 32, 32, lane);

    v8f oacc[4];                          // att tiles [c=ct*16..+15][n=n_t*16..+15]
#pragma unroll
    for (int j = 0; j < 4; ++j) oacc[j] = v8f_zero();

    for (int mc = 0; mc < 16; ++mc) {
        const int m0 = mc * 64;

        // ---- S = f^T g : one WMMA per 16x16 logit tile (K=32) ----
#pragma unroll
        for (int mi = 0; mi < 2; ++mi) {
            int m_t = grp * 2 + mi;
            v16h bf = load_frag_bt(GT + ((size_t)b * 1024 + m0 + m_t * 16) * 32,
                                   32, 0, lane);
            v8f s = wmma16(af, bf, v8f_zero());
#pragma unroll
            for (int v = 0; v < 8; ++v)
                S[n_t * 16 + v + 8 * half][m_t * 16 + nl] = s[v];
        }
        __syncthreads();

        // ---- streaming softmax, 4 threads per query row ----
        {
            float mx = -1e30f;
#pragma unroll
            for (int i = 0; i < 16; ++i) mx = fmaxf(mx, S[r][seg * 16 + i]);
            PM[r][seg] = mx;
        }
        __syncthreads();
        float newM = fmaxf(fmaxf(fmaxf(PM[r][0], PM[r][1]),
                                 fmaxf(PM[r][2], PM[r][3])), Mrun[r]);
        {
            float sum = 0.0f;
#pragma unroll
            for (int i = 0; i < 16; ++i) {
                float e = __expf(S[r][seg * 16 + i] - newM);
                P[r][seg * 16 + i] = (_Float16)e;
                sum += e;
            }
            PS[r][seg] = sum;
        }
        __syncthreads();
        if (seg == 0) {
            float corr = __expf(Mrun[r] - newM);
            Srun[r] = Srun[r] * corr + PS[r][0] + PS[r][1] + PS[r][2] + PS[r][3];
            Mrun[r] = newM;
            Cfac[r] = corr;
        }
        __syncthreads();

        // ---- rescale running accumulator, then att += h . P^T ----
        float cf = Cfac[n_t * 16 + nl];
#pragma unroll
        for (int j = 0; j < 4; ++j)
#pragma unroll
            for (int v = 0; v < 8; ++v) oacc[j][v] *= cf;

#pragma unroll
        for (int j = 0; j < 4; ++j) {
            int ct = grp * 4 + j;
            const _Float16* hb = Hm + ((size_t)b * 128 + ct * 16) * 1024 + m0;
#pragma unroll
            for (int ks = 0; ks < 2; ++ks) {            // K = 64 keys / 32
                v16h a  = load_frag_a(hb + ks * 32, 1024, lane);
                v16h bp = load_frag_bt(&P[n_t * 16][0], 72, ks * 32, lane);
                oacc[j] = wmma16(a, bp, oacc[j]);
            }
        }
        __syncthreads();   // protect S/P before next chunk overwrites them
    }

    // ---- normalize by softmax sum, stage att^T (f16) for output GEMM ----
    if (tid < 64) Inv[tid] = 1.0f / Srun[tid];
    __syncthreads();
    float inv = Inv[n_t * 16 + nl];
#pragma unroll
    for (int j = 0; j < 4; ++j) {
        int ct = grp * 4 + j;
        v8h o;
#pragma unroll
        for (int v = 0; v < 8; ++v) o[v] = (_Float16)(oacc[j][v] * inv);
        *(v8h*)(&ATT[n_t * 16 + nl][ct * 16 + 8 * half]) = o;  // c contiguous in v
    }
    __syncthreads();

    // ---- o = wo . att (K=128), out = gamma*o + x ----
    // Wave owns oc-tiles {w, w+8} x all 4 n-tiles: ATT B-frags reused across
    // both oc-tiles -> 6 loads per 8 WMMAs.
    const float gm = *gamma_p;
    v8f facc[2][4];
#pragma unroll
    for (int o2 = 0; o2 < 2; ++o2)
#pragma unroll
        for (int nt = 0; nt < 4; ++nt) facc[o2][nt] = v8f_zero();

#pragma unroll
    for (int ks = 0; ks < 4; ++ks) {                     // K = 128 / 32
        v16h a0 = load_frag_a(WO + (size_t)(w * 16) * 128 + ks * 32, 128, lane);
        v16h a1 = load_frag_a(WO + (size_t)((w + 8) * 16) * 128 + ks * 32, 128, lane);
#pragma unroll
        for (int nt = 0; nt < 4; ++nt) {
            v16h bt = load_frag_bt(&ATT[nt * 16][0], 136, ks * 32, lane);
            facc[0][nt] = wmma16(a0, bt, facc[0][nt]);
            facc[1][nt] = wmma16(a1, bt, facc[1][nt]);
        }
    }
#pragma unroll
    for (int o2 = 0; o2 < 2; ++o2)
#pragma unroll
        for (int nt = 0; nt < 4; ++nt)
#pragma unroll
            for (int v = 0; v < 8; ++v) {
                int oc = (w + o2 * 8) * 16 + v + 8 * half;
                int n  = n0 + nt * 16 + nl;
                size_t idx = ((size_t)b * 256 + oc) * 4096 + n;
                out[idx] = gm * facc[o2][nt][v] + x[idx];
            }
}

// ---------------------------------------------------------------------------
// Launch
// ---------------------------------------------------------------------------
extern "C" void kernel_launch(void* const* d_in, const int* in_sizes, int n_in,
                              void* d_out, int out_size, void* d_ws, size_t ws_size,
                              hipStream_t stream) {
    const float* x  = (const float*)d_in[0];
    const float* wf = (const float*)d_in[1];
    const float* wg = (const float*)d_in[2];
    const float* wh = (const float*)d_in[3];
    const float* wo = (const float*)d_in[4];
    const float* gp = (const float*)d_in[5];

    char* ws = (char*)d_ws;
    _Float16* XT   = (_Float16*)(ws + 0);          // 16*4096*256*2 = 33554432
    _Float16* FT   = (_Float16*)(ws + 33554432);   // 16*4096*32*2  =  4194304
    _Float16* GT   = (_Float16*)(ws + 37748736);   // 16*1024*32*2  =  1048576
    _Float16* Hm   = (_Float16*)(ws + 38797312);   // 16*128*1024*2 =  4194304
    _Float16* WF16 = (_Float16*)(ws + 42991616);   // 32*256*2
    _Float16* WG16 = (_Float16*)(ws + 43008000);   // 32*256*2
    _Float16* WH16 = (_Float16*)(ws + 43024384);   // 128*256*2
    _Float16* WO16 = (_Float16*)(ws + 43089920);   // 256*128*2

    cvt_f32_to_f16<<<32,  256, 0, stream>>>(wf, WF16, 32 * 256);
    cvt_f32_to_f16<<<32,  256, 0, stream>>>(wg, WG16, 32 * 256);
    cvt_f32_to_f16<<<128, 256, 0, stream>>>(wh, WH16, 128 * 256);
    cvt_f32_to_f16<<<128, 256, 0, stream>>>(wo, WO16, 256 * 128);

    xt_kernel<<<dim3(64, 4, 16), 256, 0, stream>>>(x, XT);

    proj_kernel<<<dim3(32, 16), 384, 0, stream>>>(XT, WF16, WG16, WH16, FT, GT, Hm);

    attn_kernel<<<dim3(64, 16), 256, 0, stream>>>(FT, GT, Hm, WO16, x, gp,
                                                  (float*)d_out);
}
